// Quantizer_35184372088910
// MI455X (gfx1250) — matrix operand order repaired and checked by reading, not compiled, not run on hardware
//
#include <hip/hip_runtime.h>

typedef __attribute__((ext_vector_type(16))) _Float16     v16h;
typedef __attribute__((ext_vector_type(8)))  float        v8f;
typedef __attribute__((ext_vector_type(4)))  unsigned int u32x4;
typedef __attribute__((ext_vector_type(8)))  int          i32x8;
typedef __attribute__((ext_vector_type(4)))  int          i32x4;

#define N_TOK 65536          // 16 * 4096 tokens
#define EDIM  64
#define N_E   1024
#define NTILE (N_E / 16)     // 64 column tiles
#define ROWS_PER_BLOCK 128   // 8 waves * 16 rows
#define BETA  0.25f
#define LDS_EMB_BYTES (N_E * EDIM * 2)          // 128 KB f16 codebook
#define LDS_TOTAL     (LDS_EMB_BYTES + N_E * 4) // + 4 KB ||e||^2

// ---------------- one-time prep: f32 codebook -> f16 + ||e||^2, zero loss ----
__global__ __launch_bounds__(256)
void vq_prep(const float* __restrict__ emb, _Float16* __restrict__ embH,
             float* __restrict__ enorm, float* __restrict__ out_loss)
{
    const int n = blockIdx.x * 256 + threadIdx.x;     // one codebook row / thread
    if (n == 0) *out_loss = 0.0f;
    const float* er = emb + (size_t)n * EDIM;
    _Float16*    dr = embH + (size_t)n * EDIM;
    float s = 0.0f;
    #pragma unroll
    for (int k = 0; k < EDIM; k += 4) {
        float4 e4 = *(const float4*)(er + k);
        s += e4.x * e4.x + e4.y * e4.y + e4.z * e4.z + e4.w * e4.w;
        dr[k + 0] = (_Float16)e4.x; dr[k + 1] = (_Float16)e4.y;
        dr[k + 2] = (_Float16)e4.z; dr[k + 3] = (_Float16)e4.w;
    }
    enorm[n] = s;
}

// ---------------- TDM: DMA a [rows x 64] f16 tile global -> LDS --------------
__device__ __forceinline__
void tdm_load_codebook(unsigned lds_off, const _Float16* gsrc, int rows)
{
    const unsigned long long ga = (unsigned long long)(const void*)gsrc;
    u32x4 g0;
    g0[0] = 1u;                                        // count=1 (valid D#)
    g0[1] = lds_off;                                   // lds_addr (bytes)
    g0[2] = (unsigned)(ga & 0xFFFFFFFFu);              // global_addr[31:0]
    g0[3] = (unsigned)((ga >> 32) & 0x01FFFFFFu)       // global_addr[56:32]
          | 0x80000000u;                               // type=2 ("image")
    i32x8 g1;
    g1[0] = 0x00010000;                                // data_size=1 -> 2 bytes
    g1[1] = (int)((unsigned)EDIM << 16);               // tensor_dim0 = 64
    g1[2] = (int)((unsigned)N_E  << 16);               // tensor_dim1 = 1024
    g1[3] = (int)((unsigned)EDIM << 16);               // tile_dim0   = 64
    g1[4] = rows;                                      // tile_dim1, tile_dim2=0
    g1[5] = EDIM;                                      // tensor_dim0_stride = 64
    g1[6] = 0;
    g1[7] = 0;
    i32x4 z4 = {0, 0, 0, 0};
#if __clang_major__ >= 23
    i32x8 z8 = {0, 0, 0, 0, 0, 0, 0, 0};
    __builtin_amdgcn_tensor_load_to_lds(g0, g1, z4, z4, z8, 0);
#else
    __builtin_amdgcn_tensor_load_to_lds(g0, g1, z4, z4, 0);
#endif
}

// ---------------- main quantizer ---------------------------------------------
__global__ __launch_bounds__(256)
void vq_quantize_kernel(const float* __restrict__ z,
                        const float* __restrict__ emb,
                        const _Float16* __restrict__ embH,
                        const float* __restrict__ enorm,
                        float* __restrict__ out_loss,
                        float* __restrict__ out_zq,
                        float* __restrict__ out_idx)
{
    extern __shared__ char smemRaw[];
    _Float16* sEmb = (_Float16*)smemRaw;                  // [1024][64] f16
    float*    sEn  = (float*)(smemRaw + LDS_EMB_BYTES);   // [1024] f32

    const int tid  = threadIdx.x;
    const int lane = tid & 31;
    const int wave = tid >> 5;
    const int rowTile = blockIdx.x * ROWS_PER_BLOCK + wave * 16;

    // ---- stage codebook: one TDM descriptor per wave (128 rows each) ----
    {
        const unsigned ldsBase = (unsigned)(size_t)(void*)sEmb; // low 32b = LDS offset
        tdm_load_codebook(ldsBase + (unsigned)(wave * 128 * EDIM * 2),
                          embH + (size_t)wave * 128 * EDIM, 128);
        // ||e||^2 staged with plain loads while the TDM runs
        float4 e4 = *(const float4*)(enorm + tid * 4);
        *(float4*)(sEn + tid * 4) = e4;
    }

    // ---- A fragments: this wave's 16x64 tile of z, f32 -> f16 ----
    // A layout (16x32): lanes 0-15 hold row M=lane, K {0..15};
    // lanes 16-31 hold row M=lane-16, K {16..31} (matches B's contiguous K order).
    const int aRow = rowTile + (lane & 15);
    const int kOff = (lane >> 4) * 16;
    const float* zr = z + (size_t)aRow * EDIM;
    v16h afrag[2];
    #pragma unroll
    for (int kb = 0; kb < 2; ++kb) {
        #pragma unroll
        for (int i = 0; i < 16; ++i)
            afrag[kb][i]     = (_Float16)zr[kb * 32 + kOff + i];
    }

    __builtin_amdgcn_s_wait_tensorcnt(0);
    __syncthreads();

    // ---- per-lane running argmin over all 64 column tiles, B double-buffered --
    float bestD[8];
    int   bestI[8];
    #pragma unroll
    for (int r = 0; r < 8; ++r) { bestD[r] = 3.4e38f; bestI[r] = 0; }

    const int nSub = lane & 15;
    const int kSel = (lane >> 4) * 16;  // B layout: lane group selects K half
    const _Float16* bBase = &sEmb[nSub * EDIM + kSel];

    v16h  b0 = *(const v16h*)(bBase);
    v16h  b1 = *(const v16h*)(bBase + 32);
    float en = sEn[nSub];

    for (int c = 0; c < NTILE; ++c) {
        // prefetch next tile's B fragments + ||e||^2 (wrap keeps it branch-free)
        const int cn = (c + 1) & (NTILE - 1);
        const v16h  nb0 = *(const v16h*)(bBase + cn * 16 * EDIM);
        const v16h  nb1 = *(const v16h*)(bBase + cn * 16 * EDIM + 32);
        const float nen = sEn[cn * 16 + nSub];

        v8f acc = {};
        acc = __builtin_amdgcn_wmma_f32_16x16x32_f16(
                  false, afrag[0], false, b0, (short)0, acc, false, false);
        acc = __builtin_amdgcn_wmma_f32_16x16x32_f16(
                  false, afrag[1], false, b1, (short)0, acc, false, false);

        const int nGlob = c * 16 + nSub;
        #pragma unroll
        for (int r = 0; r < 8; ++r) {
            const float sc = en - 2.0f * acc[r];  // ||e||^2 - 2 z.e (||z||^2 const)
            if (sc < bestD[r]) { bestD[r] = sc; bestI[r] = nGlob; }
        }
        b0 = nb0; b1 = nb1; en = nen;
    }

    // ---- argmin reduce across the 16-lane N-group (first-index tie-break) ----
    #pragma unroll
    for (int off = 1; off < 16; off <<= 1) {
        #pragma unroll
        for (int r = 0; r < 8; ++r) {
            const float od = __shfl_xor(bestD[r], off, 32);
            const int   oi = __shfl_xor(bestI[r], off, 32);
            if (od < bestD[r] || (od == bestD[r] && oi < bestI[r])) {
                bestD[r] = od; bestI[r] = oi;
            }
        }
    }

    // ---- emit indices (rows 0..7 in lanes 0-15, rows 8..15 in lanes 16-31) ----
    const int grp = lane >> 4;
    #pragma unroll
    for (int r = 0; r < 8; ++r) {
        if ((lane & 15) == r) {
            out_idx[rowTile + grp * 8 + r] = (float)bestI[r];
        }
    }

    // ---- gather z_q rows (exact f32 codebook) + loss: 2 floats/lane/row ----
    float lossAcc = 0.0f;
    #pragma unroll
    for (int m = 0; m < 16; ++m) {
        const int idx = __shfl(bestI[m & 7], (m >> 3) * 16, 32); // broadcast winner
        const int row = rowTile + m;
        const float2 zv = *(const float2*)(z   + (size_t)row * EDIM + lane * 2);
        const float2 ev = *(const float2*)(emb + (size_t)idx * EDIM + lane * 2);
        *(float2*)(out_zq + (size_t)row * EDIM + lane * 2) = ev;  // straight-through
        const float d0 = ev.x - zv.x, d1 = ev.y - zv.y;
        lossAcc += d0 * d0 + d1 * d1;
    }
    #pragma unroll
    for (int off = 16; off >= 1; off >>= 1)
        lossAcc += __shfl_xor(lossAcc, off, 32);
    if (lane == 0) {
        const float scale = (1.0f + BETA) / ((float)N_TOK * (float)EDIM);
        atomicAdd(out_loss, lossAcc * scale);
    }
}

extern "C" void kernel_launch(void* const* d_in, const int* in_sizes, int n_in,
                              void* d_out, int out_size, void* d_ws, size_t ws_size,
                              hipStream_t stream) {
    (void)in_sizes; (void)n_in; (void)out_size; (void)ws_size;
    const float* z   = (const float*)d_in[0];
    const float* emb = (const float*)d_in[1];
    float* out      = (float*)d_out;
    float* out_loss = out;                                // [1]
    float* out_zq   = out + 1;                            // [16*4096*64]
    float* out_idx  = out + 1 + (size_t)N_TOK * EDIM;     // [65536]

    _Float16* embH  = (_Float16*)d_ws;                    // 128 KB f16 codebook
    float*    enorm = (float*)((char*)d_ws + LDS_EMB_BYTES); // 4 KB ||e||^2

    vq_prep<<<N_E / 256, 256, 0, stream>>>(emb, embH, enorm, out_loss);
    vq_quantize_kernel<<<N_TOK / ROWS_PER_BLOCK, 256, LDS_TOTAL, stream>>>(
        z, emb, embH, enorm, out_loss, out_zq, out_idx);
}
